// Convolution_2d_22308060135987
// MI455X (gfx1250) — matrix operand order repaired and checked by reading, not compiled, not run on hardware
//
#include <hip/hip_runtime.h>
#include <hip/hip_bf16.h>

// Conv2d 3x3 'same' via implicit GEMM on CDNA5 WMMA (split-f16 emulation of f32).
// M = B*H*W = 131072, N = 256, K = 128*9 = 1152.
// ws layout (f16 elements): [Xhi NHWC][Xlo NHWC][Whi swizzled][Wlo swizzled]

typedef __attribute__((ext_vector_type(8)))  _Float16 v8h;
typedef __attribute__((ext_vector_type(16))) _Float16 v16h;
typedef __attribute__((ext_vector_type(8)))  float    v8f;

#define NB   32
#define CIN  128
#define COUT 256
#define HH   64
#define WW   64

static constexpr size_t X_ELEMS  = (size_t)NB * HH * WW * CIN;     // 16,777,216
static constexpr size_t WSW_ELEMS = (size_t)9 * 4 * 16 * 32 * 16;  // 294,912
static constexpr size_t XHI_OFF = 0;
static constexpr size_t XLO_OFF = X_ELEMS;
static constexpr size_t WHI_OFF = 2 * X_ELEMS;
static constexpr size_t WLO_OFF = 2 * X_ELEMS + WSW_ELEMS;

// ---------------------------------------------------------------------------
// Prepass 1: X NCHW f32 -> NHWC f16 hi/lo split, LDS-tiled transpose.
// One block per (b,h): 64 w * 128 c tile. Both global sides coalesced.
// ---------------------------------------------------------------------------
__global__ void __launch_bounds__(256) x_split_kernel(
    const float* __restrict__ X, _Float16* __restrict__ Xhi, _Float16* __restrict__ Xlo)
{
    __shared__ _Float16 shi[64 * 132];   // pad stride 132 to dodge bank conflicts
    __shared__ _Float16 slo[64 * 132];
    const int t  = threadIdx.x;
    const int bh = blockIdx.x;           // 0 .. 2047
    const int b  = bh >> 6;
    const int h  = bh & 63;

    {   // read NCHW coalesced (consecutive threads -> consecutive w)
        const int w  = t & 63;
        const int cq = t >> 6;           // 0..3
        for (int c = cq; c < CIN; c += 4) {
            float x = X[(((size_t)b * CIN + c) * HH + h) * WW + w];
            _Float16 hi = (_Float16)x;
            _Float16 lo = (_Float16)(x - (float)hi);
            shi[w * 132 + c] = hi;
            slo[w * 132 + c] = lo;
        }
    }
    __syncthreads();
    {   // write NHWC coalesced (consecutive threads -> consecutive c)
        const int c  = t & 127;
        const int wq = t >> 7;           // 0..1
        for (int w = wq; w < WW; w += 2) {
            size_t o = (((size_t)b * HH + h) * WW + w) * CIN + c;
            Xhi[o] = shi[w * 132 + c];
            Xlo[o] = slo[w * 132 + c];
        }
    }
}

// ---------------------------------------------------------------------------
// Prepass 2: W OIHW f32 -> per-lane-swizzled B fragments, hi/lo split.
// Dest element (((f*4+cb)*16 + oct)*32 + lane)*16 + j holds
//   W[oc = oct*16 + (lane&15)][ic = cb*32 + (lane>>4)*16 + j][ky][kx], f = ky*3+kx.
// B operand in this code's contiguous K order (lane = N, k = 16*(lane>>4) + j), matching A.
// ---------------------------------------------------------------------------
__global__ void __launch_bounds__(256) w_swizzle_kernel(
    const float* __restrict__ Wt, _Float16* __restrict__ Whi, _Float16* __restrict__ Wlo)
{
    const int t = blockIdx.x * 256 + threadIdx.x;
    if (t >= (int)WSW_ELEMS) return;
    const int j    = t & 15;
    const int lane = (t >> 4) & 31;
    const int oct  = (t >> 9) & 15;
    const int cb   = (t >> 13) & 3;
    const int f    = t >> 15;            // 0..8
    const int n    = lane & 15;
    const int ic   = cb * 32 + ((lane >> 4) << 4) + j;
    const int oc   = oct * 16 + n;
    const int ky   = f / 3, kx = f % 3;
    float w = Wt[((size_t)(oc * CIN + ic) * 3 + ky) * 3 + kx];
    _Float16 hi = (_Float16)w;
    _Float16 lo = (_Float16)(w - (float)hi);
    Whi[t] = hi;
    Wlo[t] = lo;
}

// ---------------------------------------------------------------------------
// Main kernel: each wave computes a 32(M) x 64(N) output tile.
// 2 M-tiles (16 consecutive w each) x 4 N-tiles (16 oc each).
// K loop: 9 filter taps x 4 channel-blocks of 32 -> 36 K-chunks,
// 3 WMMAs per (Mtile,Ntile,chunk): AhiBhi + AhiBlo + AloBhi.
// ---------------------------------------------------------------------------
__global__ void __launch_bounds__(256) conv_wmma_kernel(
    const _Float16* __restrict__ Xhi, const _Float16* __restrict__ Xlo,
    const _Float16* __restrict__ Whi, const _Float16* __restrict__ Wlo,
    const float* __restrict__ bias, float* __restrict__ out)
{
    const int lane = threadIdx.x & 31;
    const int gw   = blockIdx.x * 8 + (threadIdx.x >> 5);   // global wave id, 0..16383
    const int noc  = gw & 3;          // which 64-wide oc slab
    const int m0   = (gw >> 2) << 5;  // first of 32 consecutive pixels (one row span)
    const int w0   = m0 & 63;         // 0 or 32
    const int h    = (m0 >> 6) & 63;
    const int b    = m0 >> 12;

    const int mlane = lane & 15;      // M (A rows) / N (B,C cols)
    const int khalf = lane >> 4;      // which K half this lane holds
    const int kb8   = khalf * 8;

    v8f acc[2][4];
#pragma unroll
    for (int i = 0; i < 2; ++i)
#pragma unroll
        for (int j = 0; j < 4; ++j) { v8f z = {}; acc[i][j] = z; }

    for (int f = 0; f < 9; ++f) {
        const int ky  = f / 3, kx = f % 3;
        const int hin = h + ky - 1;
        const bool hok = (unsigned)hin < (unsigned)HH;
        for (int cb = 0; cb < 4; ++cb) {
            const int c0 = cb * 32;
            // ---- A fragments: 16x32 f16, lane=M, channels contiguous in NHWC ----
            v16h ahi[2], alo[2];
#pragma unroll
            for (int mt = 0; mt < 2; ++mt) {
                const int win = w0 + mt * 16 + mlane + kx - 1;
                v8h h0 = {}, h1 = {}, l0 = {}, l1 = {};
                if (hok && (unsigned)win < (unsigned)WW) {
                    size_t base = (((size_t)b * HH + hin) * WW + win) * CIN + c0 + khalf * 16;
                    h0 = *(const v8h*)(Xhi + base);        // K = 16*khalf..16*khalf+7
                    h1 = *(const v8h*)(Xhi + base + 8);    // K = 16*khalf+8..16*khalf+15
                    l0 = *(const v8h*)(Xlo + base);
                    l1 = *(const v8h*)(Xlo + base + 8);
                }
                ahi[mt] = __builtin_shufflevector(h0, h1, 0,1,2,3,4,5,6,7,8,9,10,11,12,13,14,15);
                alo[mt] = __builtin_shufflevector(l0, l1, 0,1,2,3,4,5,6,7,8,9,10,11,12,13,14,15);
            }
            // ---- B fragments per N tile (pre-swizzled, L2-resident) + WMMAs ----
#pragma unroll
            for (int nt = 0; nt < 4; ++nt) {
                const int oct = noc * 4 + nt;
                size_t boff = (((size_t)(f * 4 + cb) * 16 + oct) * 32 + lane) * 16;
                v8h bh0 = *(const v8h*)(Whi + boff);
                v8h bh1 = *(const v8h*)(Whi + boff + 8);
                v8h bl0 = *(const v8h*)(Wlo + boff);
                v8h bl1 = *(const v8h*)(Wlo + boff + 8);
                v16h bhi = __builtin_shufflevector(bh0, bh1, 0,1,2,3,4,5,6,7,8,9,10,11,12,13,14,15);
                v16h blo = __builtin_shufflevector(bl0, bl1, 0,1,2,3,4,5,6,7,8,9,10,11,12,13,14,15);
#pragma unroll
                for (int mt = 0; mt < 2; ++mt) {
                    acc[mt][nt] = __builtin_amdgcn_wmma_f32_16x16x32_f16(
                        false, ahi[mt], false, bhi, (short)0, acc[mt][nt], false, false);
                    acc[mt][nt] = __builtin_amdgcn_wmma_f32_16x16x32_f16(
                        false, ahi[mt], false, blo, (short)0, acc[mt][nt], false, false);
                    acc[mt][nt] = __builtin_amdgcn_wmma_f32_16x16x32_f16(
                        false, alo[mt], false, bhi, (short)0, acc[mt][nt], false, false);
                }
            }
        }
    }

    // ---- epilogue: C/D layout lane=N, VGPR r -> M = r + khalf*8 (consecutive w) ----
    const int oc_base = noc * 64;
#pragma unroll
    for (int nt = 0; nt < 4; ++nt) {
        const int oc = oc_base + nt * 16 + mlane;
        const float bn = bias[oc];
#pragma unroll
        for (int mt = 0; mt < 2; ++mt) {
            const int wb = w0 + mt * 16 + khalf * 8;
            size_t o = (((size_t)b * COUT + oc) * HH + h) * WW + wb;
            v8f c = acc[mt][nt];
            float4 v0 = make_float4(c[0] + bn, c[1] + bn, c[2] + bn, c[3] + bn);
            float4 v1 = make_float4(c[4] + bn, c[5] + bn, c[6] + bn, c[7] + bn);
            *(float4*)(out + o)     = v0;
            *(float4*)(out + o + 4) = v1;
        }
    }
}

// ---------------------------------------------------------------------------
extern "C" void kernel_launch(void* const* d_in, const int* in_sizes, int n_in,
                              void* d_out, int out_size, void* d_ws, size_t ws_size,
                              hipStream_t stream)
{
    const float* X    = (const float*)d_in[0];   // [32,128,64,64]
    const float* Wt   = (const float*)d_in[1];   // [256,128,3,3]
    const float* bias = (const float*)d_in[2];   // [256]
    float* out = (float*)d_out;                  // [32,256,64,64]

    _Float16* ws  = (_Float16*)d_ws;
    _Float16* Xhi = ws + XHI_OFF;
    _Float16* Xlo = ws + XLO_OFF;
    _Float16* Whi = ws + WHI_OFF;
    _Float16* Wlo = ws + WLO_OFF;

    x_split_kernel  <<<NB * HH, 256, 0, stream>>>(X, Xhi, Xlo);                 // 2048 blocks
    w_swizzle_kernel<<<(int)(WSW_ELEMS / 256), 256, 0, stream>>>(Wt, Whi, Wlo); // 1152 blocks
    conv_wmma_kernel<<<2048, 256, 0, stream>>>(Xhi, Xlo, Whi, Wlo, bias, out);  // 16384 waves
}
